// StreamModule_52140902974261
// MI455X (gfx1250) — compile-verified, matrix operand order checked by reading, not run on hardware
//
#include <hip/hip_runtime.h>
#include <hip/hip_bf16.h>

// ---- problem constants (match reference) ----
#define B_  8
#define T_  1024
#define F_  1024
#define C_  100
#define H_  256
#define BT_ (B_ * T_)

// ---- WMMA vector types (gfx1250, wave32) ----
typedef __attribute__((ext_vector_type(16))) __bf16 v16bf;
typedef __attribute__((ext_vector_type(8)))  __bf16 v8bf;
typedef __attribute__((ext_vector_type(8)))  float  v8f;

#define LDA 40   // halfs per LDS row: 80B -> 16B aligned frags, conflict-free walk

__device__ __forceinline__ __bf16 f2bf(float f) {
    union { float f; unsigned u; } v; v.f = f;
    unsigned r = v.u + 0x7FFFu + ((v.u >> 16) & 1u);   // round-to-nearest-even
    unsigned short h = (unsigned short)(r >> 16);
    return __builtin_bit_cast(__bf16, h);
}

__device__ __forceinline__ float f4get(const float4& v, int e) {
    switch (e) { case 0: return v.x; case 1: return v.y; case 2: return v.z; default: return v.w; }
}

struct alignas(8) bf4 { __bf16 a, b, c, d; };

// Assemble one 16x32 bf16 fragment for this lane from an LDS tile stored
// [row][k] with row-stride LDA halfs. Per ISA 7.12.2 (16-bit A 16x32):
//   lanes 0-15 : elems 0..7 = K 0..7,  elems 8..15 = K 16..23
//   lanes 16-31: elems 0..7 = K 8..15, elems 8..15 = K 24..31
// => two contiguous 16B chunks at +khalf*8 and +16+khalf*8 halfs.
__device__ __forceinline__ v16bf load_frag(const __bf16* rowp, int khalf) {
    const __bf16* p = rowp + khalf * 8;
    v8bf lo = *(const v8bf*)(p);
    v8bf hi = *(const v8bf*)(p + 16);
    union { v16bf v; v8bf h[2]; } u;
    u.h[0] = lo; u.h[1] = hi;
    return u.v;
}

// ============================================================================
// Kernel 1: for one class c and a 128-row bt tile, compute
//   h = relu(x * W1[c] + b1[c])  (128 x 256, bf16 WMMA, f32 accum, in-regs)
//   att[bt, c] = sigmoid(h . w2[c] + b2[c])
// grid = (BT/128, C), block = 256 threads (8 wave32)
// Ping-pong LDS + register prefetch: global-load latency hidden behind WMMA.
// ============================================================================
__global__ __launch_bounds__(256) void mlp_att_kernel(
    const float* __restrict__ x,   // (B*T, F)
    const float* __restrict__ W1,  // (C, F, H)
    const float* __restrict__ b1,  // (C, H)
    const float* __restrict__ w2,  // (C, H)
    const float* __restrict__ b2,  // (C,)
    float* __restrict__ att)       // (B*T, C)
{
    const int cls = blockIdx.y;
    const int bt0 = blockIdx.x * 128;
    const int tid = threadIdx.x;

    __shared__ __align__(16) __bf16 ldsA[2][128 * LDA];  // x tile    [m][k]
    __shared__ __align__(16) __bf16 ldsB[2][256 * LDA];  // W1 tile^T [h][k]
    __shared__ float scr[128 * 4];                       // cross-wave partials

    const int lane  = tid & 31;
    const int wid   = tid >> 5;
    const int wm    = wid & 1;    // 2 row-groups of 64
    const int wn    = wid >> 1;   // 4 col-groups of 64
    const int lrow  = lane & 15;
    const int khalf = lane >> 4;

    v8f acc[4][4];
#pragma unroll
    for (int mi = 0; mi < 4; ++mi)
#pragma unroll
        for (int ni = 0; ni < 4; ++ni)
            acc[mi][ni] = (v8f){0.f, 0.f, 0.f, 0.f, 0.f, 0.f, 0.f, 0.f};

    const float* w1c = W1 + (size_t)cls * F_ * H_;

    // prefetch registers for the next k-step
    float4 pa[4];        // A: 4 float4 per thread (128x32 tile)
    float4 pb[2][4];     // B: 2 units x 4 k-rows (32x256 tile)

    auto issue_loads = [&](int k) {
#pragma unroll
        for (int j = 0; j < 4; ++j) {
            int v  = tid + j * 256;      // 0..1023
            int r  = v >> 3;             // row 0..127
            int c4 = v & 7;              // float4 within row
            pa[j] = *(const float4*)(x + (size_t)(bt0 + r) * F_ + k + c4 * 4);
        }
#pragma unroll
        for (int u = 0; u < 2; ++u) {
            int uu  = tid + u * 256;     // 0..511
            int h4  = uu & 63;           // h-group of 4 (coalesced across lanes)
            int kk0 = (uu >> 6) * 4;     // k-quad 0,4,...,28
#pragma unroll
            for (int i = 0; i < 4; ++i)
                pb[u][i] = *(const float4*)(w1c + (size_t)(k + kk0 + i) * H_ + h4 * 4);
        }
    };

    auto store_stage = [&](int buf) {
#pragma unroll
        for (int j = 0; j < 4; ++j) {
            int v  = tid + j * 256;
            int r  = v >> 3;
            int c4 = v & 7;
            bf4 s; s.a = f2bf(pa[j].x); s.b = f2bf(pa[j].y);
                   s.c = f2bf(pa[j].z); s.d = f2bf(pa[j].w);
            *(bf4*)(&ldsA[buf][r * LDA + c4 * 4]) = s;
        }
#pragma unroll
        for (int u = 0; u < 2; ++u) {
            int uu  = tid + u * 256;
            int h4  = uu & 63;
            int kk0 = (uu >> 6) * 4;
            // transpose 4x4 in registers -> 4x ds_store_b64 (8B aligned)
#pragma unroll
            for (int e = 0; e < 4; ++e) {
                bf4 s;
                s.a = f2bf(f4get(pb[u][0], e));
                s.b = f2bf(f4get(pb[u][1], e));
                s.c = f2bf(f4get(pb[u][2], e));
                s.d = f2bf(f4get(pb[u][3], e));
                *(bf4*)(&ldsB[buf][(h4 * 4 + e) * LDA + kk0]) = s;
            }
        }
    };

    auto compute = [&](int buf) {
        v16bf a[4], b[4];
#pragma unroll
        for (int mi = 0; mi < 4; ++mi)
            a[mi] = load_frag(&ldsA[buf][(wm * 64 + mi * 16 + lrow) * LDA], khalf);
#pragma unroll
        for (int ni = 0; ni < 4; ++ni)
            b[ni] = load_frag(&ldsB[buf][(wn * 64 + ni * 16 + lrow) * LDA], khalf);
#pragma unroll
        for (int mi = 0; mi < 4; ++mi)
#pragma unroll
            for (int ni = 0; ni < 4; ++ni)
                acc[mi][ni] = __builtin_amdgcn_wmma_f32_16x16x32_bf16(
                    false, a[mi], false, b[ni], (short)0, acc[mi][ni], false, false);
    };

    // ---- software-pipelined main loop over K = F ----
    issue_loads(0);
    store_stage(0);
    __syncthreads();
    for (int ks = 0; ks < F_ / 32; ++ks) {
        const int kn = (ks + 1) * 32;
        const bool more = kn < F_;
        if (more) issue_loads(kn);       // global loads in flight during WMMA
        compute(ks & 1);
        if (more) store_stage((ks + 1) & 1);
        __syncthreads();
    }

    // ---- epilogue: bias + relu + dot(w2) fused, never materialize h ----
    // C/D layout: VGPR r -> row (r + 8*khalf), col = lrow within the 16x16 tile.
    float p[4][8];
#pragma unroll
    for (int mi = 0; mi < 4; ++mi)
#pragma unroll
        for (int r = 0; r < 8; ++r) p[mi][r] = 0.f;

#pragma unroll
    for (int ni = 0; ni < 4; ++ni) {
        int n = wn * 64 + ni * 16 + lrow;
        float b1v = b1[cls * H_ + n];
        float w2v = w2[cls * H_ + n];
#pragma unroll
        for (int mi = 0; mi < 4; ++mi)
#pragma unroll
            for (int r = 0; r < 8; ++r) {
                float h = acc[mi][ni][r] + b1v;
                h = h > 0.f ? h : 0.f;
                p[mi][r] = fmaf(h, w2v, p[mi][r]);
            }
    }
    // reduce over the 16 columns held by lanes {0..15} / {16..31}
#pragma unroll
    for (int mask = 1; mask < 16; mask <<= 1)
#pragma unroll
        for (int mi = 0; mi < 4; ++mi)
#pragma unroll
            for (int r = 0; r < 8; ++r)
                p[mi][r] += __shfl_xor(p[mi][r], mask, 32);

    if (lrow == 0) {
#pragma unroll
        for (int mi = 0; mi < 4; ++mi)
#pragma unroll
            for (int r = 0; r < 8; ++r) {
                int row = wm * 64 + mi * 16 + khalf * 8 + r;
                scr[row * 4 + wn] = p[mi][r];
            }
    }
    __syncthreads();

    if (tid < 128) {
        float s = scr[tid * 4 + 0] + scr[tid * 4 + 1] +
                  scr[tid * 4 + 2] + scr[tid * 4 + 3] + b2[cls];
        float a = 1.f / (1.f + __expf(-s));
        att[(size_t)(bt0 + tid) * C_ + cls] = a;
    }
}

// ============================================================================
// Kernel 2: weighted_features[b,c,f] = sum_t att[b,t,c] * x[b,t,f]
// grid = (F/128, B, 2 class-halves), block = 128; x read once per class-half.
// (1.7 GFLOP, ~0.4% of total work -> VALU FMA is the right tool; deterministic)
// ============================================================================
__global__ __launch_bounds__(128) void wf_kernel(
    const float* __restrict__ x,    // (B*T, F)
    const float* __restrict__ att,  // (B*T, C)
    float* __restrict__ out)        // (B, C, F)
{
    const int b  = blockIdx.y;
    const int f  = blockIdx.x * 128 + threadIdx.x;
    const int c0 = blockIdx.z * 50;

    __shared__ float sa[32 * 50];
    float acc[50];
#pragma unroll
    for (int i = 0; i < 50; ++i) acc[i] = 0.f;

    for (int t0 = 0; t0 < T_; t0 += 32) {
        __syncthreads();
        for (int j = threadIdx.x; j < 32 * 50; j += 128) {
            int tt = j / 50, cc = j % 50;
            sa[j] = att[(size_t)(b * T_ + t0 + tt) * C_ + c0 + cc];
        }
        __syncthreads();
        for (int tt = 0; tt < 32; ++tt) {
            float xv = x[(size_t)(b * T_ + t0 + tt) * F_ + f];
#pragma unroll
            for (int cc = 0; cc < 50; ++cc)
                acc[cc] = fmaf(sa[tt * 50 + cc], xv, acc[cc]);
        }
    }
#pragma unroll
    for (int cc = 0; cc < 50; ++cc)
        out[((size_t)b * C_ + c0 + cc) * F_ + f] = acc[cc];
}

extern "C" void kernel_launch(void* const* d_in, const int* in_sizes, int n_in,
                              void* d_out, int out_size, void* d_ws, size_t ws_size,
                              hipStream_t stream) {
    const float* x  = (const float*)d_in[0];  // (B,T,F)
    const float* W1 = (const float*)d_in[1];  // (C,F,H)
    const float* b1 = (const float*)d_in[2];  // (C,H)
    const float* w2 = (const float*)d_in[3];  // (C,H)
    const float* b2 = (const float*)d_in[4];  // (C,)

    float* wf  = (float*)d_out;                        // (B,C,F) first output
    float* att = (float*)d_out + (size_t)B_ * C_ * F_; // (B,T,C) second output

    dim3 g1(BT_ / 128, C_);
    mlp_att_kernel<<<g1, 256, 0, stream>>>(x, W1, b1, w2, b2, att);

    dim3 g2(F_ / 128, B_, 2);
    wf_kernel<<<g2, 128, 0, stream>>>(x, att, wf);
}
